// DeviceAwareAttention_13932873908580
// MI455X (gfx1250) — matrix operand order repaired and checked by reading, not compiled, not run on hardware
//
#include <hip/hip_runtime.h>
#include <hip/hip_bf16.h>

#define NUM_HEADS    32
#define NUM_KV_HEADS 8
#define HEAD_DIM     128
#define SEQ          2048
#define DMODEL       4096
#define KVDIM        1024
#define QKVN         (DMODEL + 2 * KVDIM)   /* 6144 */

typedef __attribute__((ext_vector_type(16))) __bf16 v16bf;
typedef __attribute__((ext_vector_type(8)))  float  v8f;
typedef __attribute__((ext_vector_type(4)))  float  f32x4;
typedef __attribute__((ext_vector_type(4)))  int    i32x4;

__device__ __forceinline__ unsigned short f2bf(float f) {
    unsigned u = __float_as_uint(f);
    u += 0x7fffu + ((u >> 16) & 1u);          // round-to-nearest-even
    return (unsigned short)(u >> 16);
}
__device__ __forceinline__ float bf2f(unsigned short h) {
    return __uint_as_float(((unsigned)h) << 16);
}
__device__ __forceinline__ v16bf frag16(const unsigned short* p0, const unsigned short* p1) {
    union { f32x4 f[2]; v16bf v; } u;
    u.f[0] = *(const f32x4*)p0;
    u.f[1] = *(const f32x4*)p1;
    return u.v;
}
__device__ __forceinline__ v8f wmma_bf16(v16bf a, v16bf b, v8f c) {
    return __builtin_amdgcn_wmma_f32_16x16x32_bf16(false, a, false, b, (short)0, c,
                                                   false, false);
}

// ---- CDNA5 async global->LDS path (ASYNCcnt), guarded for toolchain support
#if defined(__has_builtin)
#if __has_builtin(__builtin_amdgcn_global_load_async_to_lds_b128)
#define HAVE_ASYNC_LDS 1
#endif
#endif

#ifdef HAVE_ASYNC_LDS
typedef __attribute__((address_space(1))) i32x4 g_i32x4;   // global int4
typedef __attribute__((address_space(3))) i32x4 l_i32x4;   // LDS int4
__device__ __forceinline__ void async_b128(const void* g, void* l) {
    __builtin_amdgcn_global_load_async_to_lds_b128(
        (g_i32x4*)(unsigned long long)g,
        (l_i32x4*)(unsigned)(unsigned long long)l,
        0, 0);
}
#if __has_builtin(__builtin_amdgcn_s_wait_asynccnt)
#define WAIT_ASYNC(n) __builtin_amdgcn_s_wait_asynccnt(n)
#else
#define WAIT_ASYNC(n) asm volatile("s_wait_asynccnt %0" ::"i"(n) : "memory")
#endif
#endif

// ---------------------------------------------------------------- converts
__global__ __launch_bounds__(256)
void cvt_f32_to_bf16(const float* __restrict__ in, unsigned short* __restrict__ out, int n) {
    int i = blockIdx.x * 256 + threadIdx.x;
    if (i < n) out[i] = f2bf(in[i]);
}

// fp32 [K][N] row-major -> bf16 [N][K] row-major, tiled 32x32 through LDS
// (coalesced reads AND writes; naive version would fetch one cacheline/element)
__global__ __launch_bounds__(256)
void transpose_f32_to_bf16_tiled(const float* __restrict__ in,
                                 unsigned short* __restrict__ out, int K, int N) {
    __shared__ unsigned short tile[32][34];
    const int tx = threadIdx.x & 31;
    const int ty = threadIdx.x >> 5;          // 0..7
    const int n0 = blockIdx.x * 32;
    const int k0 = blockIdx.y * 32;
#pragma unroll
    for (int i = 0; i < 4; ++i) {
        const int k = k0 + ty + i * 8;
        tile[ty + i * 8][tx] = f2bf(in[(size_t)k * N + n0 + tx]);
    }
    __syncthreads();
#pragma unroll
    for (int i = 0; i < 4; ++i) {
        const int nrow = ty + i * 8;
        out[(size_t)(n0 + nrow) * K + k0 + tx] = tile[tx][nrow];
    }
}

// bf16 V slice of QKV [S][HEAD_DIM] (row stride QKVN) -> Vt [kv][HEAD_DIM][S]
__global__ __launch_bounds__(256)
void transpose_v_bf16(const unsigned short* __restrict__ QKV,
                      unsigned short* __restrict__ Vt) {
    __shared__ unsigned short tile[32][34];
    const int tx  = threadIdx.x & 31;
    const int ty  = threadIdx.x >> 5;
    const int s0  = blockIdx.x * 32;
    const int d0  = blockIdx.y * 32;
    const int kvh = blockIdx.z;
    const unsigned short* src = QKV + DMODEL + KVDIM + kvh * HEAD_DIM;
#pragma unroll
    for (int i = 0; i < 4; ++i)
        tile[ty + i * 8][tx] = src[(size_t)(s0 + ty + i * 8) * QKVN + d0 + tx];
    __syncthreads();
    unsigned short* dst = Vt + (size_t)kvh * HEAD_DIM * SEQ;
#pragma unroll
    for (int i = 0; i < 4; ++i) {
        const int drow = ty + i * 8;
        dst[(size_t)(d0 + drow) * SEQ + s0 + tx] = tile[tx][drow];
    }
}

// -------------------------------------------------- WMMA GEMM  C = A * B^T
// A: bf16 [M][K] row-major, Bt: bf16 [N][K] row-major, C: f32 or bf16 [M][N]
// Block tile 128x256, K step 32, 8 waves (2x4), wave tile 64x64 (16 WMMA/step).
// Double-buffered LDS; staging via GLOBAL_LOAD_ASYNC_TO_LDS_B128 when available.
__global__ __launch_bounds__(256)
void gemm_bf16_wmma(const unsigned short* __restrict__ A,
                    const unsigned short* __restrict__ Bt,
                    void* __restrict__ Cout,
                    int M, int N, int K, int c_is_f32) {
    __shared__ unsigned short sA[2][128][48];   // 12 KB per buffer
    __shared__ unsigned short sB[2][256][48];   // 24 KB per buffer

    const int t    = threadIdx.x;
    const int lane = t & 31;
    const int wave = t >> 5;
    const int bm   = blockIdx.y * 128;
    const int bn   = blockIdx.x * 256;
    const int wm   = (wave & 1) * 64;           // 2 waves along M
    const int wn   = (wave >> 1) * 64;          // 4 waves along N
    const int r    = lane & 15;
    const int half = lane >> 4;

    // staging: 2 x b128 for A (row t/2, k chunk (t&1)*16), 4 x b128 for B (row t)
    const int aRow = t >> 1;
    const int aK   = (t & 1) * 16;
    const int bRow = t;
    const unsigned short* gA = A  + (size_t)(bm + aRow) * K + aK;
    const unsigned short* gB = Bt + (size_t)(bn + bRow) * K;

    v8f acc[4][4];
#pragma unroll
    for (int mi = 0; mi < 4; ++mi)
#pragma unroll
        for (int ni = 0; ni < 4; ++ni)
#pragma unroll
            for (int j = 0; j < 8; ++j) acc[mi][ni][j] = 0.0f;

    const int steps = K / 32;

    auto stage = [&](int buf, int it) {
        const int kb = it * 32;
#ifdef HAVE_ASYNC_LDS
        async_b128(gA + kb,     &sA[buf][aRow][aK]);
        async_b128(gA + kb + 8, &sA[buf][aRow][aK + 8]);
#pragma unroll
        for (int c = 0; c < 4; ++c)
            async_b128(gB + kb + c * 8, &sB[buf][bRow][c * 8]);
#else
        f32x4 a0 = *(const f32x4*)(gA + kb);
        f32x4 a1 = *(const f32x4*)(gA + kb + 8);
        f32x4 b0 = *(const f32x4*)(gB + kb);
        f32x4 b1 = *(const f32x4*)(gB + kb + 8);
        f32x4 b2 = *(const f32x4*)(gB + kb + 16);
        f32x4 b3 = *(const f32x4*)(gB + kb + 24);
        *(f32x4*)&sA[buf][aRow][aK]     = a0;
        *(f32x4*)&sA[buf][aRow][aK + 8] = a1;
        *(f32x4*)&sB[buf][bRow][0]      = b0;
        *(f32x4*)&sB[buf][bRow][8]      = b1;
        *(f32x4*)&sB[buf][bRow][16]     = b2;
        *(f32x4*)&sB[buf][bRow][24]     = b3;
#endif
    };

    auto compute = [&](int buf) {
        v16bf afr[4];
#pragma unroll
        for (int mi = 0; mi < 4; ++mi)
            afr[mi] = frag16(&sA[buf][wm + mi * 16 + r][half * 8],
                             &sA[buf][wm + mi * 16 + r][16 + half * 8]);
#pragma unroll
        for (int ni = 0; ni < 4; ++ni) {
            v16bf bfr = frag16(&sB[buf][wn + ni * 16 + r][half * 8],
                               &sB[buf][wn + ni * 16 + r][16 + half * 8]);
#pragma unroll
            for (int mi = 0; mi < 4; ++mi)
                acc[mi][ni] = wmma_bf16(afr[mi], bfr, acc[mi][ni]);
        }
    };

    stage(0, 0);
    for (int it = 0; it + 1 < steps; ++it) {     // steady state: branch-free body
        stage((it + 1) & 1, it + 1);
#ifdef HAVE_ASYNC_LDS
        WAIT_ASYNC(6);                           // newest 6 (next buf) stay in flight
#endif
        __syncthreads();
        compute(it & 1);
        __syncthreads();
    }
#ifdef HAVE_ASYNC_LDS
    WAIT_ASYNC(0);                               // peeled last iteration
#endif
    __syncthreads();
    compute((steps - 1) & 1);

#pragma unroll
    for (int mi = 0; mi < 4; ++mi)
#pragma unroll
        for (int ni = 0; ni < 4; ++ni)
#pragma unroll
            for (int j = 0; j < 8; ++j) {
                int row = bm + wm + mi * 16 + j + half * 8;
                int col = bn + wn + ni * 16 + r;
                if (c_is_f32)
                    ((float*)Cout)[(size_t)row * N + col] = acc[mi][ni][j];
                else
                    ((unsigned short*)Cout)[(size_t)row * N + col] = f2bf(acc[mi][ni][j]);
            }
    (void)M;
}

// ---------------------------------------------- RoPE + head-major re-layout
// QKV bf16 [S][6144] -> Qr [32][S][128], Kr [8][S][128]   (V handled separately)
__global__ __launch_bounds__(64)
void rope_split(const unsigned short* __restrict__ QKV,
                unsigned short* __restrict__ Qr,
                unsigned short* __restrict__ Kr) {
    const int s = blockIdx.x, y = blockIdx.y, d = threadIdx.x;   // d in [0,64)
    const float ang = (float)s * __expf((float)d * -0.14391156855162092f); // ln(1e4)/64
    const float cs = __cosf(ang), sn = __sinf(ang);
    const unsigned short* rowp = QKV + (size_t)s * QKVN;
    if (y < NUM_HEADS) {
        const int cb = y * HEAD_DIM;
        const float x1 = bf2f(rowp[cb + d]), x2 = bf2f(rowp[cb + d + 64]);
        unsigned short* q = Qr + ((size_t)y * SEQ + s) * HEAD_DIM;
        q[d]      = f2bf(x1 * cs - x2 * sn);
        q[d + 64] = f2bf(x2 * cs + x1 * sn);
    } else {
        const int kvh = y - NUM_HEADS;
        const int cb  = DMODEL + kvh * HEAD_DIM;
        const float x1 = bf2f(rowp[cb + d]), x2 = bf2f(rowp[cb + d + 64]);
        unsigned short* k = Kr + ((size_t)kvh * SEQ + s) * HEAD_DIM;
        k[d]      = f2bf(x1 * cs - x2 * sn);
        k[d + 64] = f2bf(x2 * cs + x1 * sn);
    }
}

// ------------------------------------------------- causal flash attention
// One wave per (16-query block, head). S tile 16x32 via WMMA, online softmax,
// P re-laid through LDS into an A-fragment, O += P*V via WMMA.
__global__ __launch_bounds__(32)
void flash_attn_wmma(const unsigned short* __restrict__ Qr,
                     const unsigned short* __restrict__ Kr,
                     const unsigned short* __restrict__ Vt,
                     unsigned short* __restrict__ Attn) {
    __shared__ unsigned short pLDS[16][48];
    const int lane = threadIdx.x;
    const int q0   = blockIdx.x * 16;
    const int h    = blockIdx.y;
    const int kvh  = h >> 2;                         // N_REP = 4
    const int r    = lane & 15;
    const int half = lane >> 4;

    const unsigned short* Q  = Qr + (size_t)h   * SEQ * HEAD_DIM;
    const unsigned short* Kp = Kr + (size_t)kvh * SEQ * HEAD_DIM;
    const unsigned short* Vp = Vt + (size_t)kvh * HEAD_DIM * SEQ;

    v16bf qf[4];
#pragma unroll
    for (int dk = 0; dk < 4; ++dk) {
        const unsigned short* base = Q + (size_t)(q0 + r) * HEAD_DIM + dk * 32;
        qf[dk] = frag16(base + half * 8, base + 16 + half * 8);
    }

    v8f o[8];
    float m[8], l[8];
#pragma unroll
    for (int j = 0; j < 8; ++j) {
        m[j] = -3.0e38f; l[j] = 0.0f;
#pragma unroll
        for (int dt = 0; dt < 8; ++dt) o[dt][j] = 0.0f;
    }
    const float scale = 0.08838834764831845f;        // 1/sqrt(128)

    for (int kb = 0; kb * 32 < q0 + 16; ++kb) {      // causal bound on key blocks
        v8f s[2];
#pragma unroll
        for (int nt = 0; nt < 2; ++nt) {
#pragma unroll
            for (int j = 0; j < 8; ++j) s[nt][j] = 0.0f;
            const int key = kb * 32 + nt * 16 + r;
#pragma unroll
            for (int dk = 0; dk < 4; ++dk) {
                const unsigned short* kbase =
                    Kp + (size_t)key * HEAD_DIM + dk * 32 + half * 8;
                v16bf kf = frag16(kbase, kbase + 16);
                s[nt] = wmma_bf16(qf[dk], kf, s[nt]);
            }
        }

        float p0[8], p1[8];
#pragma unroll
        for (int j = 0; j < 8; ++j) {
            const int row = q0 + j + half * 8;
            float v0 = s[0][j] * scale; if (kb * 32 + r      > row) v0 = -3.0e38f;
            float v1 = s[1][j] * scale; if (kb * 32 + 16 + r > row) v1 = -3.0e38f;
            float mx = fmaxf(v0, v1);
#pragma unroll
            for (int off = 1; off < 16; off <<= 1)
                mx = fmaxf(mx, __shfl_xor(mx, off, 32));
            const float newm = fmaxf(m[j], mx);
            const float f    = __expf(m[j] - newm);
            p0[j] = __expf(v0 - newm);
            p1[j] = __expf(v1 - newm);
            float sum = p0[j] + p1[j];
#pragma unroll
            for (int off = 1; off < 16; off <<= 1)
                sum += __shfl_xor(sum, off, 32);
            l[j] = l[j] * f + sum;
            m[j] = newm;
#pragma unroll
            for (int dt = 0; dt < 8; ++dt) o[dt][j] *= f;
        }

        __syncthreads();                              // protect pLDS reuse
#pragma unroll
        for (int j = 0; j < 8; ++j) {
            const int prow = j + half * 8;
            pLDS[prow][r]      = f2bf(p0[j]);
            pLDS[prow][16 + r] = f2bf(p1[j]);
        }
        __syncthreads();
        v16bf pf = frag16(&pLDS[r][half * 8], &pLDS[r][16 + half * 8]);

#pragma unroll
        for (int dt = 0; dt < 8; ++dt) {
            const unsigned short* vbase =
                Vp + (size_t)(dt * 16 + r) * SEQ + kb * 32 + half * 8;
            v16bf vf = frag16(vbase, vbase + 16);
            o[dt] = wmma_bf16(pf, vf, o[dt]);
        }
    }

#pragma unroll
    for (int j = 0; j < 8; ++j) {
        const float inv = 1.0f / l[j];
        const int row = q0 + j + half * 8;
#pragma unroll
        for (int dt = 0; dt < 8; ++dt)
            Attn[(size_t)row * DMODEL + h * HEAD_DIM + dt * 16 + r] =
                f2bf(o[dt][j] * inv);
    }
}

// ------------------------------------------------------------------ launch
extern "C" void kernel_launch(void* const* d_in, const int* in_sizes, int n_in,
                              void* d_out, int out_size, void* d_ws, size_t ws_size,
                              hipStream_t stream) {
    (void)in_sizes; (void)n_in; (void)out_size; (void)ws_size;
    const float* hs = (const float*)d_in[0];
    const float* Wq = (const float*)d_in[1];
    const float* Wk = (const float*)d_in[2];
    const float* Wv = (const float*)d_in[3];
    const float* Wo = (const float*)d_in[4];

    unsigned char* ws = (unsigned char*)d_ws;
    size_t off = 0;
    auto alloc = [&](size_t bytes) {
        unsigned short* p = (unsigned short*)(ws + off);
        off += (bytes + 255) & ~(size_t)255;
        return p;
    };
    unsigned short* hsb  = alloc((size_t)SEQ * DMODEL * 2);                   // 16 MiB
    unsigned short* Wqkv = alloc((size_t)QKVN * DMODEL * 2);                  // 48 MiB
    unsigned short* Wot  = alloc((size_t)DMODEL * DMODEL * 2);                // 32 MiB
    unsigned short* QKV  = alloc((size_t)SEQ * QKVN * 2);                     // 24 MiB
    unsigned short* Qrb  = alloc((size_t)NUM_HEADS * SEQ * HEAD_DIM * 2);     // 16 MiB
    unsigned short* Krb  = alloc((size_t)NUM_KV_HEADS * SEQ * HEAD_DIM * 2);  //  4 MiB
    unsigned short* Vtb  = alloc((size_t)NUM_KV_HEADS * HEAD_DIM * SEQ * 2);  //  4 MiB
    unsigned short* Attn = alloc((size_t)SEQ * DMODEL * 2);                   // 16 MiB

    { int n = SEQ * DMODEL;
      cvt_f32_to_bf16<<<(n + 255) / 256, 256, 0, stream>>>(hs, hsb, n); }

    transpose_f32_to_bf16_tiled<<<dim3(DMODEL / 32, DMODEL / 32), 256, 0, stream>>>(
        Wq, Wqkv, DMODEL, DMODEL);
    transpose_f32_to_bf16_tiled<<<dim3(KVDIM / 32, DMODEL / 32), 256, 0, stream>>>(
        Wk, Wqkv + (size_t)DMODEL * DMODEL, DMODEL, KVDIM);
    transpose_f32_to_bf16_tiled<<<dim3(KVDIM / 32, DMODEL / 32), 256, 0, stream>>>(
        Wv, Wqkv + (size_t)(DMODEL + KVDIM) * DMODEL, DMODEL, KVDIM);
    transpose_f32_to_bf16_tiled<<<dim3(DMODEL / 32, DMODEL / 32), 256, 0, stream>>>(
        Wo, Wot, DMODEL, DMODEL);

    gemm_bf16_wmma<<<dim3(QKVN / 256, SEQ / 128), 256, 0, stream>>>(
        hsb, Wqkv, QKV, SEQ, QKVN, DMODEL, 0);

    rope_split<<<dim3(SEQ, NUM_HEADS + NUM_KV_HEADS), 64, 0, stream>>>(QKV, Qrb, Krb);
    transpose_v_bf16<<<dim3(SEQ / 32, HEAD_DIM / 32, NUM_KV_HEADS), 256, 0, stream>>>(
        QKV, Vtb);

    flash_attn_wmma<<<dim3(SEQ / 16, NUM_HEADS), 32, 0, stream>>>(Qrb, Krb, Vtb, Attn);

    gemm_bf16_wmma<<<dim3(DMODEL / 256, SEQ / 128), 256, 0, stream>>>(
        Attn, Wot, d_out, SEQ, DMODEL, DMODEL, 1);
}
